// ContextualAttentionModule_59889023975725
// MI455X (gfx1250) — compile-verified, matrix operand order checked by reading, not run on hardware
//
#include <hip/hip_runtime.h>
#include <hip/hip_bf16.h>
#include <stdint.h>

// ---------------------------------------------------------------------------
// CDNA5 WMMA types
// ---------------------------------------------------------------------------
typedef __attribute__((ext_vector_type(16))) __bf16 v16bf;
typedef __attribute__((ext_vector_type(8)))  float  v8f;

union Frag {
    uint4 u[2];   // 32 bytes = 16 bf16
    v16bf v;
};

__device__ __forceinline__ unsigned short f2bf(float f) {
    // round-to-nearest-even fp32 -> bf16
    unsigned int u = __float_as_uint(f);
    unsigned int r = (u + 0x7FFFu + ((u >> 16) & 1u)) >> 16;
    return (unsigned short)r;
}

// ---------------------------------------------------------------------------
// Prep: per spatial position x (= patch l), build
//   Afm[x][cd]  = im2col(foreground)            (bf16)
//   Kmat[l][cd] = normalized im2col(background) (bf16)   (+1e-7 everywhere)
//   KT[cd][l]   = Kmat transposed               (bf16)
// cd = c*27 + (a*9 + b*3 + e),  offset = (a-1, b-1, e-1)
// ---------------------------------------------------------------------------
__global__ __launch_bounds__(256)
void prep_kernel(const float* __restrict__ fg, const float* __restrict__ mask,
                 unsigned short* __restrict__ Afm, unsigned short* __restrict__ Kmat,
                 unsigned short* __restrict__ KT)
{
    const int x   = blockIdx.x;          // 0..4095
    const int tid = threadIdx.x;
    const int xi = x >> 8, xj = (x >> 4) & 15, xk = x & 15;

    float bgv_loc[7];
    float ssq = 0.f;
    #pragma unroll
    for (int it = 0; it < 7; ++it) {
        int cd = tid + it * 256;
        float bgv = 0.f;
        if (cd < 1728) {
            int c = cd / 27, d = cd % 27;
            int a = d / 9, b = (d / 3) % 3, e = d % 3;
            int yi = xi + a - 1, yj = xj + b - 1, yk = xk + e - 1;
            bool inb = (unsigned)yi < 16u && (unsigned)yj < 16u && (unsigned)yk < 16u;
            float fmv = 0.f;
            if (inb) {
                int y = (yi << 8) | (yj << 4) | yk;
                fmv = fg[c * 4096 + y];
                bgv = fmv * (1.0f - mask[y]);
            }
            bgv += 1e-7f;                      // reference adds eps to every entry
            Afm[(size_t)x * 1728 + cd] = f2bf(fmv);
            ssq += bgv * bgv;
        }
        bgv_loc[it] = bgv;
    }

    __shared__ float red[256];
    red[tid] = ssq;
    __syncthreads();
    for (int s = 128; s > 0; s >>= 1) {
        if (tid < s) red[tid] += red[tid + s];
        __syncthreads();
    }
    const float rnorm = rsqrtf(red[0]);

    #pragma unroll
    for (int it = 0; it < 7; ++it) {
        int cd = tid + it * 256;
        if (cd < 1728) {
            unsigned short bv = f2bf(bgv_loc[it] * rnorm);
            Kmat[(size_t)x * 1728 + cd] = bv;
            KT[(size_t)cd * 4096 + x]   = bv;
        }
    }
}

// ---------------------------------------------------------------------------
// C[M x N] = A[M x K] * B[N x K]^T   (bf16 in, f32 accumulate via WMMA)
//
// Block = 128 threads = 4 waves stacked in M (256 rows/block); all 4 waves
// share the same 64-column N tile, so the 64x32 bf16 B tile (4KB) is staged
// into LDS with CDNA5 async copies (global_load_async_to_lds_b128, ASYNCcnt),
// double-buffered so the copy of tile k+1 overlaps the WMMAs of tile k:
//   issue copy -> s_wait_asynccnt 0 -> s_barrier -> ds_load fragments -> wmma
// A rows are wave-private -> direct global_load_b128.
//
// Fragment layout per ISA 7.12.2 (16-bit A 16x32): lane<16 -> row=lane,
// K chunks {0..7,16..23}; lane>=16 -> row=lane-16, K chunks {8..15,24..31}.
// B mirrors A with N in place of M (B supplied row-major [N x K]).
// ---------------------------------------------------------------------------
__global__ __launch_bounds__(128)
void gemm_abt_bf16(float* __restrict__ C,
                   const unsigned short* __restrict__ A,
                   const unsigned short* __restrict__ B,
                   int lda, int ldb, int ldc, int Kd)
{
    __shared__ __align__(16) unsigned short sB[2][64 * 32];   // 2 x 4KB

    const int tid  = threadIdx.x;
    const int lane = tid & 31;
    const int wave = tid >> 5;
    const int r    = lane & 15;
    const int koff = (lane >> 4) * 8;

    const int mBase = blockIdx.x * 256 + wave * 64;
    const int nBase = blockIdx.y * 64;

    // addrspace(3) offset = low 32 bits of the generic LDS address
    const unsigned ldsB0 = (unsigned)(uintptr_t)&sB[0][0];
    const unsigned ldsB1 = (unsigned)(uintptr_t)&sB[1][0];

    // async-stage one 64x32 B tile (4KB = 256 x 16B chunks, 2 per thread)
    auto copyB = [&](unsigned ldsBase, int kk) {
        #pragma unroll
        for (int q = tid; q < 256; q += 128) {
            int row = q >> 2;                 // 0..63
            int c16 = q & 3;                  // 16B chunk within the 64B row
            const char* g = (const char*)(B + (size_t)(nBase + row) * ldb + kk) + c16 * 16;
            unsigned    l = ldsBase + (unsigned)(row * 64 + c16 * 16);
            asm volatile("global_load_async_to_lds_b128 %0, %1, off"
                         :: "v"(l), "v"(g) : "memory");
        }
    };

    v8f acc[4][4];
    #pragma unroll
    for (int i = 0; i < 4; ++i)
        #pragma unroll
        for (int j = 0; j < 4; ++j)
            acc[i][j] = (v8f){};

    const unsigned short* aRow[4];
    #pragma unroll
    for (int t = 0; t < 4; ++t)
        aRow[t] = A + (size_t)(mBase + t * 16 + r) * lda + koff;

    copyB(ldsB0, 0);                          // prologue: stage first tile

    for (int k = 0; k < Kd; k += 32) {
        const int buf = (k >> 5) & 1;

        asm volatile("s_wait_asynccnt 0x0" ::: "memory");   // this wave's copies done
        __syncthreads();                                    // all waves' copies done,
                                                            // previous ds reads retired
        if (k + 32 < Kd)
            copyB(buf ? ldsB0 : ldsB1, k + 32);             // overlap next-tile copy

        Frag fa[4], fb[4];
        const unsigned short* sbase = &sB[buf][0];
        #pragma unroll
        for (int t = 0; t < 4; ++t) {
            fa[t].u[0] = *(const uint4*)(aRow[t] + k);
            fa[t].u[1] = *(const uint4*)(aRow[t] + k + 16);
            const unsigned short* bp = sbase + (t * 16 + r) * 32 + koff;
            fb[t].u[0] = *(const uint4*)(bp);
            fb[t].u[1] = *(const uint4*)(bp + 16);
        }
        if (k + 32 < Kd)
            __builtin_prefetch(aRow[0] + k + 32, 0, 1);     // global_prefetch_b8

        #pragma unroll
        for (int i = 0; i < 4; ++i)
            #pragma unroll
            for (int j = 0; j < 4; ++j)
                acc[i][j] = __builtin_amdgcn_wmma_f32_16x16x32_bf16(
                    false, fa[i].v, false, fb[j].v,
                    (short)0, acc[i][j], false, false);
    }

    // C/D layout: VGPR v -> M = v + (lane>=16 ? 8 : 0), N = lane & 15
    const int rowSel = (lane >> 4) * 8;
    const int col    = lane & 15;
    #pragma unroll
    for (int i = 0; i < 4; ++i)
        #pragma unroll
        for (int j = 0; j < 4; ++j)
            #pragma unroll
            for (int v = 0; v < 8; ++v) {
                int row = mBase + i * 16 + rowSel + v;
                C[(size_t)row * ldc + nBase + j * 16 + col] = acc[i][j][v];
            }
}

// ---------------------------------------------------------------------------
// Separable 3-wide box-sum along one spatial axis of x (rows of [4096 x 4096])
// shift = 0 (w), 4 (h), 8 (d); zero boundary (matches padded conv).
// ---------------------------------------------------------------------------
__global__ void box_pass(const float* __restrict__ src, float* __restrict__ dst, int shift)
{
    size_t id = (size_t)blockIdx.x * blockDim.x + threadIdx.x;   // 4096*4096
    int x   = (int)(id >> 12);
    int pos = (x >> shift) & 15;
    size_t off = (size_t)4096 << shift;
    float v = src[id];
    if (pos > 0)  v += src[id - off];
    if (pos < 15) v += src[id + off];
    dst[id] = v;
}

// ---------------------------------------------------------------------------
// Row-wise softmax over l (4096) per spatial x; writes bf16 attention.
// ---------------------------------------------------------------------------
__global__ __launch_bounds__(256)
void softmax_row(const float* __restrict__ P, unsigned short* __restrict__ attn)
{
    const int x   = blockIdx.x;
    const int tid = threadIdx.x;
    const float* row = P + (size_t)x * 4096;
    __shared__ float red[256];

    float m = -3.0e38f;
    for (int l = tid; l < 4096; l += 256) m = fmaxf(m, row[l]);
    red[tid] = m; __syncthreads();
    for (int s = 128; s > 0; s >>= 1) {
        if (tid < s) red[tid] = fmaxf(red[tid], red[tid + s]);
        __syncthreads();
    }
    m = red[0]; __syncthreads();

    float sum = 0.f;
    for (int l = tid; l < 4096; l += 256) sum += __expf(row[l] - m);
    red[tid] = sum; __syncthreads();
    for (int s = 128; s > 0; s >>= 1) {
        if (tid < s) red[tid] += red[tid + s];
        __syncthreads();
    }
    const float inv = 1.0f / red[0];

    unsigned short* arow = attn + (size_t)x * 4096;
    for (int l = tid; l < 4096; l += 256)
        arow[l] = f2bf(__expf(row[l] - m) * inv);
}

// ---------------------------------------------------------------------------
// Fold (col2im of the transpose-conv) + final blend:
//   recovered[c,p] = sum_delta V[p + 1 - delta, (c,delta)]
//   out = recovered*mask/27 + fg*(1-mask)
// ---------------------------------------------------------------------------
__global__ void fold_final(const float* __restrict__ V, const float* __restrict__ fg,
                           const float* __restrict__ mask, float* __restrict__ out)
{
    int tid = blockIdx.x * blockDim.x + threadIdx.x;   // 64*4096
    int c = tid >> 12;
    int x = tid & 4095;
    int xi = x >> 8, xj = (x >> 4) & 15, xk = x & 15;

    float acc = 0.f;
    #pragma unroll
    for (int a = 0; a < 3; ++a)
        #pragma unroll
        for (int b = 0; b < 3; ++b)
            #pragma unroll
            for (int e = 0; e < 3; ++e) {
                int yi = xi + 1 - a, yj = xj + 1 - b, yk = xk + 1 - e;
                if ((unsigned)yi < 16u && (unsigned)yj < 16u && (unsigned)yk < 16u) {
                    int y = (yi << 8) | (yj << 4) | yk;
                    acc += V[(size_t)y * 1728 + c * 27 + (a * 9 + b * 3 + e)];
                }
            }
    float mk = mask[x];
    out[tid] = acc * mk * (1.0f / 27.0f) + fg[tid] * (1.0f - mk);
}

// ---------------------------------------------------------------------------
// Launch: process the 4 samples sequentially on `stream`, reusing workspace.
// Workspace map (bytes):
//   [0, 14.16M)         Afm  bf16 4096x1728      \  after GEMM1 both dead ->
//   [14.16M, 28.31M)    Kmat bf16 4096x1728      /  V (f32 4096x1728) overlays
//   [28.31M, 42.47M)    KT   bf16 1728x4096
//   [42.47M, 109.6M)    Cbuf f32 4096x4096   (also reused as bf16 Attn)
//   [109.6M, 176.7M)    Pbuf f32 4096x4096
// ---------------------------------------------------------------------------
extern "C" void kernel_launch(void* const* d_in, const int* in_sizes, int n_in,
                              void* d_out, int out_size, void* d_ws, size_t ws_size,
                              hipStream_t stream)
{
    const float* fg   = (const float*)d_in[0];   // [4,64,16,16,16]
    const float* mask = (const float*)d_in[1];   // [4,1,16,16,16]
    float* out = (float*)d_out;                  // [4,64,16,16,16]

    const size_t L = 4096, CD = 1728;
    unsigned short* Afm  = (unsigned short*)d_ws;         // L*CD bf16
    unsigned short* Kmat = Afm + L * CD;                  // L*CD bf16
    unsigned short* KT   = Kmat + L * CD;                 // CD*L bf16
    float* Cbuf = (float*)(KT + CD * L);                  // L*L f32
    float* Pbuf = Cbuf + L * L;                           // L*L f32
    float* Vbuf = (float*)d_ws;                           // L*CD f32 (overlays Afm+Kmat)
    unsigned short* Attn = (unsigned short*)Cbuf;         // L*L bf16 (overlays Cbuf)

    for (int s = 0; s < 4; ++s) {
        const float* fg_s = fg   + (size_t)s * 64 * 4096;
        const float* mk_s = mask + (size_t)s * 4096;
        float*      out_s = out  + (size_t)s * 64 * 4096;

        prep_kernel<<<4096, 256, 0, stream>>>(fg_s, mk_s, Afm, Kmat, KT);
        // GEMM1: C[x,l] = Afm . Kmat^T   (M=4096, N=4096, K=1728)
        gemm_abt_bf16<<<dim3(16, 64), 128, 0, stream>>>(Cbuf, Afm, Kmat, 1728, 1728, 4096, 1728);
        // propagation box-sum, separable (C -> P -> C -> P)
        box_pass<<<65536, 256, 0, stream>>>(Cbuf, Pbuf, 0);
        box_pass<<<65536, 256, 0, stream>>>(Pbuf, Cbuf, 4);
        box_pass<<<65536, 256, 0, stream>>>(Cbuf, Pbuf, 8);
        // softmax over l -> bf16 attention
        softmax_row<<<4096, 256, 0, stream>>>(Pbuf, Attn);
        // GEMM2: V[x,cd] = Attn . KT^T   (M=4096, N=1728, K=4096)
        gemm_abt_bf16<<<dim3(16, 27), 128, 0, stream>>>(Vbuf, Attn, KT, 4096, 4096, 1728, 4096);
        fold_final<<<1024, 256, 0, stream>>>(Vbuf, fg_s, mk_s, out_s);
    }
}
